// MetaSVDD_72335839199296
// MI455X (gfx1250) — matrix operand order repaired and checked by reading, not compiled, not run on hardware
//
#include <hip/hip_runtime.h>

typedef __attribute__((ext_vector_type(16))) _Float16 v16h;
typedef __attribute__((ext_vector_type(8)))  _Float16 v8h;
typedef __attribute__((ext_vector_type(8)))  float    v8f;
typedef unsigned int uint32x4 __attribute__((ext_vector_type(4)));
typedef unsigned int uint32x8 __attribute__((ext_vector_type(8)));

#define TASKS   256
#define SHOT    32
#define NQ      64
#define INDIM   1024
#define HID     2048
#define EMB     512
#define SVDD_EPS 1e-6f
#define QP_ITERS 500

#define WMMA_F16(a, b, c) \
  __builtin_amdgcn_wmma_f32_16x16x32_f16(false, (a), false, (b), (short)0, (c), false, false)

static __device__ __forceinline__ v16h cat8(v8h lo, v8h hi) {
  return __builtin_shufflevector(lo, hi, 0,1,2,3,4,5,6,7,8,9,10,11,12,13,14,15);
}

// ---------------------------------------------------------------------------
// Kernel 0: transpose + f32->f16 convert (A[rows x cols] -> AT[cols x rows])
// so WMMA B-fragments become contiguous 32B loads.
// ---------------------------------------------------------------------------
__global__ __launch_bounds__(256)
void transpose_cvt_f16(const float* __restrict__ A, _Float16* __restrict__ AT,
                       int rows, int cols) {
  __shared__ float tile[32][33];
  const int x = blockIdx.x * 32 + threadIdx.x;   // col in A
  const int y0 = blockIdx.y * 32;
#pragma unroll
  for (int i = 0; i < 32; i += 8)
    tile[threadIdx.y + i][threadIdx.x] = A[(size_t)(y0 + threadIdx.y + i) * cols + x];
  __syncthreads();
  const int xo = y0 + threadIdx.x;               // col in AT
  const int yo0 = blockIdx.x * 32;
#pragma unroll
  for (int i = 0; i < 32; i += 8)
    AT[(size_t)(yo0 + threadIdx.y + i) * rows + xo] =
        (_Float16)tile[threadIdx.x][threadIdx.y + i];
}

// ---------------------------------------------------------------------------
// Kernel 1: fused 2-layer MLP.  One WG = 16 output rows x full 512 emb cols.
// 8 waves: phase1 builds relu(X@W1+b1) 128 hid cols at a time into LDS,
// phase2 accumulates into 32 persistent 16x16 f32 tiles (4 per wave).
// Output written as f32 (query path) and/or f16 (support path, WMMA consumer).
// ---------------------------------------------------------------------------
__global__ __launch_bounds__(256)
void fused_mlp(const float* __restrict__ X, const _Float16* __restrict__ W1T,
               const float* __restrict__ B1, const _Float16* __restrict__ W2T,
               const float* __restrict__ B2,
               float* __restrict__ OUTF, _Float16* __restrict__ OUTH) {
  __shared__ _Float16 Xs[16][INDIM + 8];   // 33 KB, row stride 16B-aligned
  __shared__ _Float16 Hs[16][128 + 8];     // 4.3 KB

  const int tid  = threadIdx.x;
  const int wid  = tid >> 5;
  const int lane = tid & 31;
  const int hl   = lane >> 4;        // half-wave select for A/B striping
  const int m    = lane & 15;
  const size_t rowbase = (size_t)blockIdx.x * 16;

  // stage X tile: 16 x 1024 f32 -> f16 LDS (coalesced float4)
  {
    const float4* Xv = (const float4*)(X + rowbase * INDIM);
#pragma unroll
    for (int i = 0; i < 16; ++i) {
      const int idx = tid + i * 256;         // 4096 float4 total
      const int r = idx >> 8;                // 256 float4 per row
      const int c = (idx & 255) << 2;
      const float4 v = Xv[idx];
      Xs[r][c + 0] = (_Float16)v.x; Xs[r][c + 1] = (_Float16)v.y;
      Xs[r][c + 2] = (_Float16)v.z; Xs[r][c + 3] = (_Float16)v.w;
    }
  }
  __syncthreads();

  v8f accs[4];
#pragma unroll
  for (int j = 0; j < 4; ++j) accs[j] = (v8f){0,0,0,0,0,0,0,0};

  for (int ch = 0; ch < HID / 128; ++ch) {
    const int hbase = ch * 128;

    // ---- phase 1: H chunk [16 x 128]; wave w -> hid cols w*16 .. w*16+15
    {
      v8f h = {0,0,0,0,0,0,0,0};
      const int col = wid * 16 + m;                       // chunk-local col (= B-frag N)
      const _Float16* w1row = W1T + (size_t)(hbase + col) * INDIM;
      for (int k = 0; k < INDIM; k += 32) {
        const int ka = k + hl * 16;
        const v16h a = cat8(*(const v8h*)&Xs[m][ka],
                            *(const v8h*)&Xs[m][ka + 8]);
        const v16h b = *(const v16h*)(w1row + k + hl * 16);
        h = WMMA_F16(a, b, h);
      }
      const float bias = B1[hbase + col];
#pragma unroll
      for (int r = 0; r < 8; ++r) {
        const int row = hl * 8 + r;                       // C-layout M
        Hs[row][col] = (_Float16)fmaxf(h[r] + bias, 0.f);
      }
    }
    __syncthreads();

    // ---- phase 2: acc[16 x 512] += Hs @ W2T-chunk; wave w -> emb cols [w*64, w*64+63]
#pragma unroll
    for (int kh = 0; kh < 128; kh += 32) {
      const int ka = kh + hl * 16;
      const v16h a = cat8(*(const v8h*)&Hs[m][ka],
                          *(const v8h*)&Hs[m][ka + 8]);
      const int kb = hbase + kh + hl * 16;
#pragma unroll
      for (int j = 0; j < 4; ++j) {
        const v16h b = *(const v16h*)(W2T + (size_t)(wid * 64 + j * 16 + m) * HID + kb);
        accs[j] = WMMA_F16(a, b, accs[j]);
      }
    }
    __syncthreads();
  }

  // epilogue: bias + store
#pragma unroll
  for (int j = 0; j < 4; ++j) {
    const int ncol = wid * 64 + j * 16 + m;
    const float bias = B2[ncol];
#pragma unroll
    for (int r = 0; r < 8; ++r) {
      const int row = hl * 8 + r;
      const float v = accs[j][r] + bias;
      if (OUTF) OUTF[(rowbase + row) * EMB + ncol] = v;
      if (OUTH) OUTH[(rowbase + row) * EMB + ncol] = (_Float16)v;
    }
  }
}

// ---------------------------------------------------------------------------
// Kernel 2: per-task SVDD dual QP + query logits.  One WG per task.
// Support tile staged LDS-padded via the Tensor Data Mover (one DMA, one
// wave issues, TENSORcnt-tracked).  K = E E^T (+eps I) via WMMA (waves 0-3),
// 500-iter exponentiated-gradient on wave 0 (K row in registers, shfl
// broadcasts, fast exp/log), then centers and -||c - q||^2 cooperatively.
// ---------------------------------------------------------------------------
__global__ __launch_bounds__(256)
void svdd_logits(const _Float16* __restrict__ SembH, const float* __restrict__ Qemb,
                 float* __restrict__ out) {
  __shared__ __align__(128) _Float16 Es[SHOT][EMB + 16];  // 33.8 KB, 32B-aligned rows
  __shared__ float Kmat[SHOT][SHOT + 1];
  __shared__ float a_lds[SHOT];
  __shared__ float rowsum[SHOT];
  __shared__ float eta_s;
  __shared__ float centers[EMB];

  const int b    = blockIdx.x;
  const int tid  = threadIdx.x;
  const int wid  = tid >> 5;
  const int lane = tid & 31;
  const int hl   = lane >> 4;
  const int m    = lane & 15;

  // ---- TDM: DMA the 32x512 f16 support tile into the padded LDS layout.
  // D# describes a 2D tile (data_size=2B, dims in elements) with LDS padding
  // of 8 DWORDs (pad_amount code 7) every 256 DWORDs (pad_interval code 7,
  // = one 1024B row) -> exactly the Es[SHOT][EMB+16] row stride.
  if (wid == 0) {
    const unsigned long long gaddr =
        (unsigned long long)(uintptr_t)(SembH + (size_t)b * SHOT * EMB);
    const unsigned ldsoff = (unsigned)(uintptr_t)&Es[0][0];
    uint32x4 g0;
    g0[0] = 1u;                                   // count=1, user descriptor
    g0[1] = ldsoff;                               // lds_addr
    g0[2] = (unsigned)(gaddr & 0xFFFFFFFFu);      // global_addr[31:0]
    g0[3] = (unsigned)((gaddr >> 32) & 0x1FFFFFFu) | (2u << 30);  // [56:32] | type=2
    uint32x8 g1;
    g1[0] = (1u << 16)        // data_size = 2 bytes
          | (1u << 20)        // pad_enable
          | (7u << 22)        // pad_interval: 256 DWORDs (one 512-elem row)
          | (7u << 25);       // pad_amount:   8 DWORDs (16 f16)
    g1[1] = (unsigned)EMB << 16;                  // tensor_dim0[15:0] @ bits 63:48
    g1[2] = (unsigned)SHOT << 16;                 // tensor_dim1[15:0] @ bits 95:80
    g1[3] = (unsigned)EMB << 16;                  // tile_dim0 @ bits 127:112
    g1[4] = (unsigned)SHOT;                       // tile_dim1 @ bits 143:128
    g1[5] = (unsigned)EMB;                        // tensor_dim0_stride[31:0]
    g1[6] = 0u;                                   // stride[47:32], dim1_stride lo
    g1[7] = 0u;
    asm volatile("tensor_load_to_lds %0, %1" :: "s"(g0), "s"(g1) : "memory");
    __builtin_amdgcn_s_wait_tensorcnt(0);
  }
  __syncthreads();

  // K = E E^T + eps I  (four 16x16 tiles, one per wave 0..3)
  if (wid < 4) {
    const int it = wid >> 1, jt = wid & 1;
    v8f acc = {0,0,0,0,0,0,0,0};
    for (int k = 0; k < EMB; k += 32) {
      const int ka = k + hl * 16;
      const v16h a = cat8(*(const v8h*)&Es[it * 16 + m][ka],
                          *(const v8h*)&Es[it * 16 + m][ka + 8]);
      const v16h bb = *(const v16h*)&Es[jt * 16 + m][k + hl * 16];  // E^T col = E row
      acc = WMMA_F16(a, bb, acc);
    }
#pragma unroll
    for (int r = 0; r < 8; ++r) {
      const int row = it * 16 + hl * 8 + r;
      const int col = jt * 16 + m;
      float v = acc[r];
      if (row == col) v += SVDD_EPS;
      Kmat[row][col] = v;
    }
  }
  __syncthreads();

  // Gershgorin step size: eta = 1 / (2 * max_s sum_t |K[s][t]|)
  if (tid < SHOT) {
    float s = 0.f;
    for (int t = 0; t < SHOT; ++t) s += fabsf(Kmat[tid][t]);
    rowsum[tid] = s;
  }
  __syncthreads();
  if (tid == 0) {
    float mx = rowsum[0];
    for (int t = 1; t < SHOT; ++t) mx = fmaxf(mx, rowsum[t]);
    eta_s = 0.5f / mx;
  }
  __syncthreads();

  // mirror descent on the simplex: wave 0, lane = sample index
  if (wid == 0) {
    float Krow[SHOT];
#pragma unroll
    for (int t = 0; t < SHOT; ++t) Krow[t] = Kmat[lane][t];
    const float kd  = Krow[lane];
    const float eta = eta_s;
    float av = 1.f / SHOT;
    for (int iter = 0; iter < QP_ITERS; ++iter) {
      float g = 0.f;
#pragma unroll
      for (int t = 0; t < SHOT; ++t) g = fmaf(Krow[t], __shfl(av, t), g);
      g = 2.f * g - kd;
      const float u = __logf(av + 1e-30f) - eta * g;
      float mx = u;
#pragma unroll
      for (int off = 16; off; off >>= 1) mx = fmaxf(mx, __shfl_xor(mx, off));
      const float e = __expf(u - mx);
      float ss = e;
#pragma unroll
      for (int off = 16; off; off >>= 1) ss += __shfl_xor(ss, off);
      av = e / ss;
    }
    a_lds[lane] = av;
  }
  __syncthreads();

  // centers = alpha^T E
  for (int d = tid; d < EMB; d += 256) {
    float c = 0.f;
#pragma unroll
    for (int s = 0; s < SHOT; ++s) c = fmaf(a_lds[s], (float)Es[s][d], c);
    centers[d] = c;
  }
  __syncthreads();

  // logits[q] = -||centers - Q[q]||^2   (4 lanes per query)
  {
    const float* Q = Qemb + (size_t)b * NQ * EMB;
    const int q = tid >> 2, p = tid & 3;
    float sum = 0.f;
    for (int d = p; d < EMB; d += 4) {
      const float diff = centers[d] - Q[(size_t)q * EMB + d];
      sum = fmaf(diff, diff, sum);
    }
    sum += __shfl_xor(sum, 1);
    sum += __shfl_xor(sum, 2);
    if (p == 0) out[(size_t)b * NQ + q] = -sum;
  }
}

// ---------------------------------------------------------------------------
// launch
// ---------------------------------------------------------------------------
extern "C" void kernel_launch(void* const* d_in, const int* in_sizes, int n_in,
                              void* d_out, int out_size, void* d_ws, size_t ws_size,
                              hipStream_t stream) {
  (void)in_sizes; (void)n_in; (void)out_size; (void)ws_size;
  const float* sup = (const float*)d_in[0];
  const float* qry = (const float*)d_in[1];
  const float* W1  = (const float*)d_in[2];
  const float* B1  = (const float*)d_in[3];
  const float* W2  = (const float*)d_in[4];
  const float* B2  = (const float*)d_in[5];
  float* out = (float*)d_out;

  char* ws = (char*)d_ws;
  _Float16* W1T   = (_Float16*)(ws);                       //  4 MB  [HID][INDIM]
  _Float16* W2T   = (_Float16*)(ws + ((size_t)4  << 20));  //  2 MB  [EMB][HID]
  _Float16* SembH = (_Float16*)(ws + ((size_t)6  << 20));  //  8 MB  [TASKS*SHOT][EMB] f16
  float*    Qemb  = (float*)   (ws + ((size_t)14 << 20));  // 32 MB  [TASKS*NQ][EMB] f32

  transpose_cvt_f16<<<dim3(HID / 32, INDIM / 32), dim3(32, 8), 0, stream>>>(W1, W1T, INDIM, HID);
  transpose_cvt_f16<<<dim3(EMB / 32, HID / 32), dim3(32, 8), 0, stream>>>(W2, W2T, HID, EMB);
  fused_mlp<<<(TASKS * SHOT) / 16, 256, 0, stream>>>(sup, W1T, B1, W2T, B2, nullptr, SembH);
  fused_mlp<<<(TASKS * NQ) / 16, 256, 0, stream>>>(qry, W1T, B1, W2T, B2, Qemb, nullptr);
  svdd_logits<<<TASKS, 256, 0, stream>>>(SembH, Qemb, out);
}
